// MultiHeadAttention_54726473285659
// MI455X (gfx1250) — matrix operand order repaired and checked by reading, not compiled, not run on hardware
//
#include <hip/hip_runtime.h>

typedef __bf16 bf16;
typedef __attribute__((ext_vector_type(16))) __bf16 v16bf;
typedef __attribute__((ext_vector_type(2)))  __bf16 bf16x2;
typedef __attribute__((ext_vector_type(8)))  float  v8f;
typedef __attribute__((ext_vector_type(4)))  float  f32x4;
typedef __attribute__((ext_vector_type(4)))  unsigned int u32x4;
typedef __attribute__((ext_vector_type(4)))  int v4i;

#define DMODEL 512
#define TSEQ   2048
#define NBATCH 4
#define NHEAD  8
#define DHEAD  64
#define MTOT   (NBATCH * TSEQ)   // 8192

// Async global->LDS copy path (CDNA5). Guarded so either toolchain compiles;
// fallback is a synchronous VGPR round-trip.
#if __has_builtin(__builtin_amdgcn_global_load_async_to_lds_b128) && \
    __has_builtin(__builtin_amdgcn_s_wait_asynccnt)
#define USE_ASYNC_LDS 1
#else
#define USE_ASYNC_LDS 0
#endif

__device__ __forceinline__ void async_copy16(void* lds, const void* g) {
#if USE_ASYNC_LDS
  // Probe-verified param 0 type: v4i in AS1 (printed as "int vector __device__*").
  __builtin_amdgcn_global_load_async_to_lds_b128(
      (__attribute__((address_space(1))) v4i*)g,
      (__attribute__((address_space(3))) v4i*)lds, 0, 0);
#else
  *(u32x4*)lds = *(const u32x4*)g;
#endif
}

__device__ __forceinline__ void async_wait_all() {
#if USE_ASYNC_LDS
  __builtin_amdgcn_s_wait_asynccnt(0);
#endif
}

__device__ __forceinline__ v8f zero_v8f() {
  v8f z;
#pragma unroll
  for (int i = 0; i < 8; ++i) z[i] = 0.0f;
  return z;
}

// A fragment (16x32 bf16, M x K) from row-major bf16 [rows][stride].
// ISA layout: row = lane%16 (both halves), lanes 16-31 take K-offset +8;
// VGPR j in 0..3 -> K = kb+2j, VGPR 4..7 -> K = 16+kb+2(j-4).
__device__ __forceinline__ v16bf load_a_frag(const bf16* p, int stride, int lane) {
  const bf16* rp = p + (size_t)(lane & 15) * stride;
  const int kb = (lane & 16) ? 8 : 0;
  v16bf a;
#pragma unroll
  for (int j = 0; j < 4; ++j) {
    bf16x2 lo = *(const bf16x2*)(rp + kb + 2 * j);
    bf16x2 hi = *(const bf16x2*)(rp + 16 + kb + 2 * j);
    a[2 * j + 0] = lo.x;     a[2 * j + 1] = lo.y;
    a[8 + 2 * j + 0] = hi.x; a[8 + 2 * j + 1] = hi.y;
  }
  return a;
}

// B fragment (32x16 bf16, K x N). Source stored n-major [n][k] with row
// stride `stride` so K-pairs are contiguous. col n = lane%16; lanes 0-15
// hold K=0..7,16..23, lanes 16-31 hold K=8..15,24..31 (2 per VGPR).
__device__ __forceinline__ v16bf load_b_frag(const bf16* p, int stride, int lane) {
  const bf16* np = p + (size_t)(lane & 15) * stride + ((lane & 16) ? 8 : 0);
  v16bf b;
#pragma unroll
  for (int j = 0; j < 8; ++j) {
    bf16x2 v = *(const bf16x2*)(np + ((j < 4) ? 2 * j : 16 + 2 * (j - 4)));
    b[2 * j + 0] = v.x;
    b[2 * j + 1] = v.y;
  }
  return b;
}

#define WMMA_BF16(A, B, C) \
  __builtin_amdgcn_wmma_f32_16x16x32_bf16(false, (A), false, (B), (short)0, (C), false, false)

// ---------------------------------------------------------------------------
// Kernel 1: fused QKV projection, fp32 -> bf16 WMMA GEMM.
// C[8192x512] = X[8192x512] @ W[512x512].  blockIdx.z selects Q / K / V.
// Workgroup: 128 threads (4 waves), 64x64 output tile, K-step 32.
// V output is stored transposed per head: Vt[n][h][d][t].
// ---------------------------------------------------------------------------
__global__ __launch_bounds__(128) void qkv_proj_kernel(
    const float* __restrict__ query, const float* __restrict__ key,
    const float* __restrict__ Wq, const float* __restrict__ Wk,
    const float* __restrict__ Wv,
    bf16* __restrict__ Qbf, bf16* __restrict__ Kbf, bf16* __restrict__ Vt) {
  __shared__ bf16 ldsX[64][32];    // X tile, row-major (A pairs contiguous)
  __shared__ bf16 ldsWt[64][32];   // W tile transposed: [n][k] (B pairs contiguous)

  const int z = blockIdx.z;
  const float* __restrict__ X = (z == 0) ? query : key;
  const float* __restrict__ W = (z == 0) ? Wq : ((z == 1) ? Wk : Wv);
  const int m0 = blockIdx.x * 64;
  const int n0 = blockIdx.y * 64;
  const int tid = threadIdx.x;
  const int lane = tid & 31;
  const int w = tid >> 5;

  v8f acc[4];
#pragma unroll
  for (int i = 0; i < 4; ++i) acc[i] = zero_v8f();

  for (int k0 = 0; k0 < DMODEL; k0 += 32) {
    // Stage X tile 64x32 (fp32 -> bf16), 4 float4 per thread.
#pragma unroll
    for (int t = 0; t < 4; ++t) {
      int i = tid + t * 128;            // 0..511
      int row = i >> 3;
      int c4 = i & 7;
      f32x4 f = *(const f32x4*)(X + (size_t)(m0 + row) * DMODEL + k0 + c4 * 4);
      bf16* dst = &ldsX[row][c4 * 4];
      dst[0] = (bf16)f.x; dst[1] = (bf16)f.y; dst[2] = (bf16)f.z; dst[3] = (bf16)f.w;
    }
    // Stage W tile 32x64 transposed into ldsWt[n][k].
    {
      const int nn = tid & 63;
      const int kk = tid >> 6;          // 0 or 1
#pragma unroll
      for (int k = kk; k < 32; k += 2)
        ldsWt[nn][k] = (bf16)W[(size_t)(k0 + k) * DMODEL + n0 + nn];
    }
    __syncthreads();

    v16bf a = load_a_frag(&ldsX[w * 16][0], 32, lane);
#pragma unroll
    for (int nt = 0; nt < 4; ++nt) {
      v16bf b = load_b_frag(&ldsWt[nt * 16][0], 32, lane);
      acc[nt] = WMMA_BF16(a, b, acc[nt]);
    }
    __syncthreads();
  }

  // Writeback (fp32 accum -> bf16). C layout: VGPR r holds row r (lanes 0-15)
  // or row r+8 (lanes 16-31); col = lane%16.
  const int rbase = (lane & 16) ? 8 : 0;
#pragma unroll
  for (int nt = 0; nt < 4; ++nt) {
#pragma unroll
    for (int r = 0; r < 8; ++r) {
      int grow = m0 + w * 16 + r + rbase;
      int gcol = n0 + nt * 16 + (lane & 15);
      bf16 val = (bf16)acc[nt][r];
      if (z == 0) {
        Qbf[(size_t)grow * DMODEL + gcol] = val;
      } else if (z == 1) {
        Kbf[(size_t)grow * DMODEL + gcol] = val;
      } else {
        int nb = grow >> 11, tt = grow & (TSEQ - 1);
        int hh = gcol >> 6, dd = gcol & 63;
        Vt[(((size_t)nb * NHEAD + hh) * DHEAD + dd) * TSEQ + tt] = val;
      }
    }
  }
}

// ---------------------------------------------------------------------------
// Kernel 2: flash attention. One workgroup = (batch, head, 64 query rows);
// 4 waves, each wave owns 16 query rows. Streams K/V in 64-key tiles that are
// double-buffered in LDS via async global->LDS DMA (ASYNCcnt), with online
// softmax (fp32 stats), QK^T and PV as bf16 WMMAs, fp32 accumulators.
// ---------------------------------------------------------------------------
__global__ __launch_bounds__(128) void flash_attn_kernel(
    const bf16* __restrict__ Qbf, const bf16* __restrict__ Kbf,
    const bf16* __restrict__ Vt, float* __restrict__ out) {
  __shared__ bf16 ldsK[2][64][64];    // K tile: [t][d]  (QK^T B pairs along d)
  __shared__ bf16 ldsV[2][64][64];    // V tile transposed: [d][t] (PV B pairs along t)
  __shared__ bf16 ldsP[4][16][64];    // per-wave P tile: [row][key]

  const int tid = threadIdx.x;
  const int lane = tid & 31;
  const int w = tid >> 5;
  const int qb = blockIdx.x * 64;
  const int n = (int)blockIdx.y >> 3;
  const int h = (int)blockIdx.y & 7;
  const float scale = 0.044194173824159216f;  // 1/sqrt(512)
  const int NT = TSEQ / 64;

  const bf16* kbase = Kbf + (size_t)(n * TSEQ) * DMODEL + h * DHEAD;
  const bf16* vbase = Vt + ((size_t)n * NHEAD + h) * DHEAD * TSEQ;

  // Stage one 64-key K tile + V^T tile into LDS buffer `buf` (4x16B each).
  auto stage_kv = [&](int buf, int kt) {
    const bf16* kgp = kbase + (size_t)(kt * 64) * DMODEL;
    const bf16* vgp = vbase + kt * 64;
#pragma unroll
    for (int t = 0; t < 4; ++t) {
      int i = tid + t * 128;            // 0..511
      int row = i >> 3;
      int q8 = i & 7;
      async_copy16(&ldsK[buf][row][q8 * 8], kgp + (size_t)row * DMODEL + q8 * 8);
      async_copy16(&ldsV[buf][row][q8 * 8], vgp + (size_t)row * TSEQ + q8 * 8);
    }
  };

  // Q fragments for this wave's 16 rows (held in registers for whole kernel).
  const bf16* qp = Qbf + (size_t)(n * TSEQ + qb + w * 16) * DMODEL + h * DHEAD;
  v16bf qa0 = load_a_frag(qp, DMODEL, lane);        // d = 0..31
  v16bf qa1 = load_a_frag(qp + 32, DMODEL, lane);   // d = 32..63

  v8f o[4];
#pragma unroll
  for (int i = 0; i < 4; ++i) o[i] = zero_v8f();
  float m_r[8], l_r[8];
#pragma unroll
  for (int r = 0; r < 8; ++r) { m_r[r] = -1.0e30f; l_r[r] = 0.0f; }

  stage_kv(0, 0);
  async_wait_all();
  __syncthreads();

  for (int kt = 0; kt < NT; ++kt) {
    const int cur = kt & 1;
    // Kick off the DMA for the next tile into the other buffer, then compute.
    if (kt + 1 < NT) stage_kv(cur ^ 1, kt + 1);

    // S = Q K^T  (4 N-tiles of 16 keys, two K-chunks of d each).
    v8f s[4];
#pragma unroll
    for (int nt = 0; nt < 4; ++nt) {
      v16bf kb0 = load_b_frag(&ldsK[cur][nt * 16][0], 64, lane);
      v8f acc = zero_v8f();
      acc = WMMA_BF16(qa0, kb0, acc);
      v16bf kb1 = load_b_frag(&ldsK[cur][nt * 16][32], 64, lane);
      s[nt] = WMMA_BF16(qa1, kb1, acc);
    }

    // Online softmax in C-fragment layout. Lane's index r corresponds to
    // query row r + (lane>=16 ? 8 : 0); a row spans a 16-lane half-group.
#pragma unroll
    for (int nt = 0; nt < 4; ++nt)
#pragma unroll
      for (int r = 0; r < 8; ++r) s[nt][r] = s[nt][r] * scale;

    float mnew[8], ef[8], rs[8];
#pragma unroll
    for (int r = 0; r < 8; ++r) {
      float v = fmaxf(fmaxf(s[0][r], s[1][r]), fmaxf(s[2][r], s[3][r]));
      v = fmaxf(v, __shfl_xor(v, 1, 32));
      v = fmaxf(v, __shfl_xor(v, 2, 32));
      v = fmaxf(v, __shfl_xor(v, 4, 32));
      v = fmaxf(v, __shfl_xor(v, 8, 32));
      mnew[r] = fmaxf(m_r[r], v);
      ef[r] = __expf(m_r[r] - mnew[r]);
      rs[r] = 0.0f;
    }

    const int prow = (lane & 16) ? 8 : 0;
    const int pcol = lane & 15;
#pragma unroll
    for (int nt = 0; nt < 4; ++nt)
#pragma unroll
      for (int r = 0; r < 8; ++r) {
        float p = __expf(s[nt][r] - mnew[r]);
        rs[r] += p;
        ldsP[w][r + prow][nt * 16 + pcol] = (bf16)p;
      }

#pragma unroll
    for (int r = 0; r < 8; ++r) {
      float v = rs[r];
      v += __shfl_xor(v, 1, 32);
      v += __shfl_xor(v, 2, 32);
      v += __shfl_xor(v, 4, 32);
      v += __shfl_xor(v, 8, 32);
      l_r[r] = l_r[r] * ef[r] + v;
      m_r[r] = mnew[r];
    }
#pragma unroll
    for (int nt = 0; nt < 4; ++nt)
#pragma unroll
      for (int r = 0; r < 8; ++r) o[nt][r] = o[nt][r] * ef[r];

    // O += P V  (wave-private P buffer; same-wave LDS ops are in order).
    v16bf pa0 = load_a_frag(&ldsP[w][0][0], 64, lane);   // keys 0..31
    v16bf pa1 = load_a_frag(&ldsP[w][0][32], 64, lane);  // keys 32..63
#pragma unroll
    for (int nt = 0; nt < 4; ++nt) {
      v16bf vb0 = load_b_frag(&ldsV[cur][nt * 16][0], 64, lane);
      o[nt] = WMMA_BF16(pa0, vb0, o[nt]);
      v16bf vb1 = load_b_frag(&ldsV[cur][nt * 16][32], 64, lane);
      o[nt] = WMMA_BF16(pa1, vb1, o[nt]);
    }

    // Next tile's DMA must be complete and all waves done reading `cur`
    // before it gets overwritten in the next iteration.
    async_wait_all();
    __syncthreads();
  }

  // Epilogue: normalize by row sums, write fp32 output [n][tq][h*64+d].
  const int rbase = (lane & 16) ? 8 : 0;
  float inv_l[8];
#pragma unroll
  for (int r = 0; r < 8; ++r) inv_l[r] = 1.0f / l_r[r];
#pragma unroll
  for (int nt = 0; nt < 4; ++nt)
#pragma unroll
    for (int r = 0; r < 8; ++r) {
      int tq = qb + w * 16 + r + rbase;
      int d = nt * 16 + (lane & 15);
      out[(size_t)(n * TSEQ + tq) * DMODEL + h * DHEAD + d] = o[nt][r] * inv_l[r];
    }
}

// ---------------------------------------------------------------------------
extern "C" void kernel_launch(void* const* d_in, const int* in_sizes, int n_in,
                              void* d_out, int out_size, void* d_ws, size_t ws_size,
                              hipStream_t stream) {
  const float* query = (const float*)d_in[0];  // [4,2048,512]
  const float* key   = (const float*)d_in[1];  // [4,2048,512]
  const float* Wq    = (const float*)d_in[2];  // [512,512]
  const float* Wk    = (const float*)d_in[3];  // [512,512]
  const float* Wv    = (const float*)d_in[4];  // [512,512]
  float* out = (float*)d_out;                  // [4,2048,512] fp32

  // Workspace: bf16 Q [8192,512], bf16 K [8192,512], bf16 V^T [4][8][64][2048]
  bf16* Qbf = (bf16*)d_ws;
  bf16* Kbf = Qbf + (size_t)MTOT * DMODEL;
  bf16* Vt  = Kbf + (size_t)MTOT * DMODEL;     // total 24 MiB

  dim3 gemm_grid(MTOT / 64, DMODEL / 64, 3);
  qkv_proj_kernel<<<gemm_grid, 128, 0, stream>>>(query, key, Wq, Wk, Wv, Qbf, Kbf, Vt);

  dim3 attn_grid(TSEQ / 64, NBATCH * NHEAD);
  flash_attn_kernel<<<attn_grid, 128, 0, stream>>>(Qbf, Kbf, Vt, out);
}
